// sqgkt_87797721465158
// MI455X (gfx1250) — matrix operand order repaired and checked by reading, not compiled, not run on hardware
//
#include <hip/hip_runtime.h>
#include <math.h>

typedef __attribute__((ext_vector_type(16))) _Float16 v16h;
typedef __attribute__((ext_vector_type(8)))  float    v8f;
typedef _Float16 half_t;

#define B_    128
#define T_    150
#define Tm1   149
#define Dm    100
#define Kp    128      // padded K for aggregate GEMM (100 -> 128)
#define KC    224      // padded K for lin/Wih GEMMs (200 -> 224)
#define MR    (B_*Tm1) // 19072 rows, multiple of 16
#define NQNB  4
#define NSNB  10
#define NUNB  10
#define NQ2NB 5
#define RANKK 10

__device__ inline float sigmoidf_(float x) { return 1.f / (1.f + expf(-x)); }

// ---- scalar row linears (f32) ----
__device__ inline void lin_rows(const float* __restrict__ W, const float* __restrict__ bias,
                                float (*S)[Dm], float (*O)[Dm], int rows, bool relu) {
  for (int o = threadIdx.x; o < rows * Dm; o += 256) {
    int r = o / Dm, n = o - r * Dm;
    float acc = bias[n];
    const float* w = W + (size_t)n * Dm;
    const float* s = S[r];
    for (int k = 0; k < Dm; ++k) acc += s[k] * w[k];
    O[r][n] = (relu && acc < 0.f) ? 0.f : acc;
  }
}
__device__ inline void lin_row(const float* __restrict__ W, const float* __restrict__ bias,
                               const float* s, float* o, bool relu) {
  int n = threadIdx.x;
  if (n < Dm) {
    float acc = bias[n];
    const float* w = W + (size_t)n * Dm;
    for (int k = 0; k < Dm; ++k) acc += s[k] * w[k];
    o[n] = (relu && acc < 0.f) ? 0.f : acc;
  }
}

// ---- triangular multi-hop aggregate (level2 via WMMA, level1/0 scalar) ----
__device__ void aggregate_one(
    int root, const int* __restrict__ NBa, int Wa, const int* __restrict__ NBb, int Wb,
    const float* __restrict__ Tev, const float* __restrict__ Tod,
    const half_t* __restrict__ WH2, const float* __restrict__ agg_W,
    const float* __restrict__ agg_b,
    const float* __restrict__ last_W, const float* __restrict__ last_b,
    half_t (*S2)[Kp], float (*OUT2)[Dm],
    float (*SCR)[Dm], float (*H1A)[Dm], float (*H1B)[Dm],
    float* VA, float* VB, float* VC, int* IDX1, int* IDX2, float* outv)
{
  const int tid = threadIdx.x, lane = tid & 31, wave = tid >> 5;
  const int R = Wa * Wb;
  const int Mt = (R + 15) >> 4;
  const float inva = 1.f / (float)Wa, invb = 1.f / (float)Wb;
  __syncthreads();
  if (tid < Wa) IDX1[tid] = NBa[(size_t)root * Wa + tid];
  __syncthreads();
  if (tid < R) IDX2[tid] = NBb[(size_t)IDX1[tid / Wb] * Wb + (tid % Wb)];
  __syncthreads();
  // S2[r] = hop2[r] + mean_j3 hop3[r][j3]   (f16, K padded to 128)
  for (int r = wave; r < Mt * 16; r += 8) {
    if (r < R) {
      const int i2 = IDX2[r];
      const float* ev = Tev + (size_t)i2 * Dm;
      const int* nb3 = NBa + (size_t)i2 * Wa;
      for (int d = lane; d < Kp; d += 32) {
        float v = 0.f;
        if (d < Dm) {
          float s = 0.f;
          for (int j3 = 0; j3 < Wa; ++j3) s += Tod[(size_t)nb3[j3] * Dm + d];
          v = ev[d] + s * inva;
        }
        S2[r][d] = (half_t)v;
      }
    } else {
      for (int d = lane; d < Kp; d += 32) S2[r][d] = (half_t)0.f;
    }
  }
  __syncthreads();
  // OUT2 = relu(S2 @ W2^T + b2) via v_wmma_f32_16x16x32_f16
  const float* b2 = agg_b + 2 * Dm;
  for (int tile = wave; tile < Mt * 7; tile += 8) {
    const int mt = tile / 7, nt = tile % 7;
    v8f acc = {};
    const int am = mt * 16 + (lane & 15);
    const int koff = (lane >> 4) << 3;        // A: half-wave K offset 0/8
    const int bn = nt * 16 + (lane & 15);
    const int bko = (lane >> 4) << 3;         // B: half-wave K offset 0/8
#pragma unroll
    for (int kt = 0; kt < Kp / 32; ++kt) {
      v16h a, bfr;
#pragma unroll
      for (int h = 0; h < 16; ++h) {
        const int v = h >> 1, hi = h & 1;
        const int k = kt * 32 + ((v >> 2) << 4) + koff + ((v & 3) << 1) + hi;
        a[h] = S2[am][k];
      }
      const half_t* bp = WH2 + (size_t)bn * Kp + kt * 32 + bko;
#pragma unroll
      for (int h = 0; h < 16; ++h) bfr[h] = bp[(h < 8) ? h : h + 8];
      acc = __builtin_amdgcn_wmma_f32_16x16x32_f16(false, a, false, bfr,
                                                   (short)0, acc, false, false);
    }
    const int n = nt * 16 + (lane & 15);
    const float bn_ = (n < Dm) ? b2[n] : 0.f;   // hoisted: one bias load per lane
#pragma unroll
    for (int v = 0; v < 8; ++v) {
      const int m = mt * 16 + v + ((lane < 16) ? 0 : 8);
      if (m < R && n < Dm) {
        float x = acc[v] + bn_;
        OUT2[m][n] = x > 0.f ? x : 0.f;
      }
    }
  }
  __syncthreads();
  // h1a = relu(lin1(mean_j2 hop2 + hop1))
  for (int o = tid; o < Wa * Dm; o += 256) {
    const int j1 = o / Dm, d = o - j1 * Dm;
    float s = 0.f;
    for (int j2 = 0; j2 < Wb; ++j2) s += Tev[(size_t)IDX2[j1 * Wb + j2] * Dm + d];
    SCR[j1][d] = Tod[(size_t)IDX1[j1] * Dm + d] + s * invb;
  }
  __syncthreads();
  lin_rows(agg_W + Dm * Dm, agg_b + Dm, SCR, H1A, Wa, true);
  __syncthreads();
  // h1b = relu(lin1(mean_j2 h2a + h1a))
  for (int o = tid; o < Wa * Dm; o += 256) {
    const int j1 = o / Dm, d = o - j1 * Dm;
    float s = 0.f;
    for (int j2 = 0; j2 < Wb; ++j2) s += OUT2[j1 * Wb + j2][d];
    SCR[j1][d] = H1A[j1][d] + s * invb;
  }
  __syncthreads();
  lin_rows(agg_W + Dm * Dm, agg_b + Dm, SCR, H1B, Wa, true);
  __syncthreads();
  // h0 chain
  for (int d = tid; d < Dm; d += 256) {
    float s = 0.f;
    for (int j1 = 0; j1 < Wa; ++j1) s += Tod[(size_t)IDX1[j1] * Dm + d];
    VA[d] = Tev[(size_t)root * Dm + d] + s * inva;
  }
  __syncthreads();
  lin_row(agg_W, agg_b, VA, VB, true);   // h0a
  __syncthreads();
  for (int d = tid; d < Dm; d += 256) {
    float s = 0.f;
    for (int j1 = 0; j1 < Wa; ++j1) s += H1A[j1][d];
    VA[d] = VB[d] + s * inva;
  }
  __syncthreads();
  lin_row(agg_W, agg_b, VA, VC, true);   // h0b
  __syncthreads();
  for (int d = tid; d < Dm; d += 256) {
    float s = 0.f;
    for (int j1 = 0; j1 < Wa; ++j1) s += H1B[j1][d];
    VA[d] = VC[d] + s * inva;
  }
  __syncthreads();
  lin_row(agg_W, agg_b, VA, VB, true);   // h0c
  __syncthreads();
  lin_row(last_W, last_b, VB, outv, true);
}

// ---- kernel: per-(b,t) aggregates -> packed f16 [emb_hat, emb_r] rows ----
__global__ void __launch_bounds__(256) k_aggregate(
    const int* __restrict__ user, const int* __restrict__ question,
    const int* __restrict__ response, const int* __restrict__ mask,
    const int* __restrict__ q_neighbors, const int* __restrict__ s_neighbors,
    const int* __restrict__ u_neighbors, const int* __restrict__ q_neighbors_2,
    const float* __restrict__ emb_q, const float* __restrict__ emb_q2,
    const float* __restrict__ emb_s, const float* __restrict__ emb_u,
    const float* __restrict__ emb_r,
    const float* __restrict__ agg_W, const float* __restrict__ agg_b,
    const float* __restrict__ last_W, const float* __restrict__ last_b,
    const float* __restrict__ w1p, const float* __restrict__ w2p,
    const half_t* __restrict__ WH2, half_t* __restrict__ XCAT)
{
  __shared__ half_t S2[64][Kp];
  __shared__ float OUT2[64][Dm];
  __shared__ float SCR[10][Dm], H1A[10][Dm], H1B[10][Dm];
  __shared__ float VA[Dm], VB[Dm], VC[Dm];
  __shared__ float EO1[Dm], EO2[Dm];
  __shared__ int IDX1[10], IDX2[64];
  const int t = blockIdx.x, b = blockIdx.y, tid = threadIdx.x;
  const int q = question[b * T_ + t];
  const int u = user[b * T_ + t];
  const int rt = response[b * T_ + t];
  const int m = mask[b * T_ + t];
  aggregate_one(q, q_neighbors, NQNB, s_neighbors, NSNB, emb_q, emb_s,
                WH2, agg_W, agg_b, last_W, last_b,
                S2, OUT2, SCR, H1A, H1B, VA, VB, VC, IDX1, IDX2, EO1);
  __syncthreads();
  aggregate_one(u, u_neighbors, NUNB, q_neighbors_2, NQ2NB, emb_u, emb_q2,
                WH2, agg_W, agg_b, last_W, last_b,
                S2, OUT2, SCR, H1A, H1B, VA, VB, VC, IDX1, IDX2, EO2);
  __syncthreads();
  const float w1 = w1p[0], w2 = w2p[0];
  const size_t rowo = ((size_t)b * Tm1 + t) * KC;
  for (int d = tid; d < KC; d += 256) {
    float v;
    if (d < Dm) {
      float eq  = m ? EO1[d] : emb_q[(size_t)q * Dm + d];
      float eq2 = m ? EO2[d] : emb_q2[(size_t)q * Dm + d];
      v = w1 * eq + w2 * eq2;
    } else if (d < 2 * Dm) {
      v = emb_r[rt * Dm + (d - Dm)];
    } else v = 0.f;
    XCAT[rowo + d] = (half_t)v;
  }
}

// ---- generic streaming WMMA GEMM: C = act(A[M,KT] @ W[N,KT]^T + bias) ----
// KT compile-time -> fully unrolled k-loop (bigger clauses, more WMMA overlap)
template <int KT>
__global__ void __launch_bounds__(256) k_gemm(
    const half_t* __restrict__ A, int lda, const half_t* __restrict__ Bw, int ldb,
    const float* __restrict__ bias, float* __restrict__ Cf, half_t* __restrict__ Ch,
    int ldc, int M, int N, int relu)
{
  const int lane = threadIdx.x & 31, wave = threadIdx.x >> 5;
  const int nt = blockIdx.x;
  const int mt = blockIdx.y * 8 + wave;
  if (mt * 16 >= M) return;                 // wave-uniform
  v8f acc = {};
  const int am = mt * 16 + (lane & 15);
  const int koff = (lane >> 4) << 3;
  const int bn = nt * 16 + (lane & 15);
  const int bko = (lane >> 4) << 3;
  const half_t* arow = A + (size_t)am * lda;
  const half_t* brow = Bw + (size_t)bn * ldb + bko;
#pragma unroll
  for (int kt = 0; kt < KT / 32; ++kt) {
    v16h a, bfr;
#pragma unroll
    for (int h = 0; h < 16; ++h) {
      const int v = h >> 1, hi = h & 1;
      const int k = kt * 32 + ((v >> 2) << 4) + koff + ((v & 3) << 1) + hi;
      a[h] = arow[k];
    }
    const half_t* bp = brow + kt * 32;
#pragma unroll
    for (int h = 0; h < 16; ++h) bfr[h] = bp[(h < 8) ? h : h + 8];
    acc = __builtin_amdgcn_wmma_f32_16x16x32_f16(false, a, false, bfr,
                                                 (short)0, acc, false, false);
  }
  const int n = nt * 16 + (lane & 15);
  const bool nin = n < N;
  const float bv = nin ? bias[n] : 0.f;       // hoisted: one bias load per lane
#pragma unroll
  for (int v = 0; v < 8; ++v) {
    const int m = mt * 16 + v + ((lane < 16) ? 0 : 8);
    float x = nin ? (acc[v] + bv) : 0.f;
    if (relu && x < 0.f) x = 0.f;
    if (Cf && nin) Cf[(size_t)m * ldc + n] = x;
    if (Ch) Ch[(size_t)m * ldc + n] = (half_t)x;
  }
}

// ---- weight f32 -> f16 padded packing ----
__global__ void k_pad_weights(const float* __restrict__ agg_W,
                              const float* __restrict__ lin_W,
                              const float* __restrict__ Wih,
                              half_t* __restrict__ WH2, half_t* __restrict__ WHLIN,
                              half_t* __restrict__ WHWIH)
{
  const int i = blockIdx.x * blockDim.x + threadIdx.x;
  const int n0 = 112 * Kp, n1 = 208 * KC, n2 = 400 * KC;
  if (i < n0) {
    int n = i / Kp, k = i % Kp;
    WH2[i] = (half_t)((n < Dm && k < Dm) ? agg_W[2 * Dm * Dm + n * Dm + k] : 0.f);
  } else if (i < n0 + n1) {
    int j = i - n0, n = j / KC, k = j % KC;
    WHLIN[j] = (half_t)((n < 2 * Dm && k < 2 * Dm) ? lin_W[n * 2 * Dm + k] : 0.f);
  } else if (i < n0 + n1 + n2) {
    int j = i - n0 - n1, n = j / KC, k = j % KC;
    WHWIH[j] = (half_t)((k < 2 * Dm) ? Wih[n * 2 * Dm + k] : 0.f);
  }
}

__global__ void k_eqseq(const int* __restrict__ question,
                        const float* __restrict__ emb_q, float* __restrict__ EQSEQ) {
  const int i = blockIdx.x * blockDim.x + threadIdx.x;
  if (i < B_ * T_ * Dm) {
    const int d = i % Dm, bt = i / Dm;
    EQSEQ[i] = emb_q[(size_t)question[bt] * Dm + d];
  }
}

__global__ void k_zero_h(half_t* __restrict__ p, size_t n) {
  for (size_t i = (size_t)blockIdx.x * blockDim.x + threadIdx.x; i < n;
       i += (size_t)gridDim.x * blockDim.x) p[i] = (half_t)0.f;
}

// ---- recap top-k + sq precompute per (b,t) ----
__global__ void __launch_bounds__(256) k_recap(
    const int* __restrict__ question, const int* __restrict__ qs_skill_ids,
    const float* __restrict__ emb_q, const float* __restrict__ emb_s,
    const float* __restrict__ qW, const float* __restrict__ qb,
    const float* __restrict__ wW,
    const float* __restrict__ EQSEQ, float* __restrict__ SQ, int* __restrict__ TOPIDX)
{
  __shared__ float eqn[Dm];
  __shared__ float sc[T_];
  __shared__ float qsr[5][Dm];
  __shared__ float part[5][Dm];
  const int t = blockIdx.x, b = blockIdx.y, tid = threadIdx.x;
  const size_t row = (size_t)b * Tm1 + t;
  const float* eb = EQSEQ + ((size_t)b * T_ + (t + 1)) * Dm;
  if (tid < Dm) eqn[tid] = eb[tid];
  __syncthreads();
  for (int j = tid; j < T_; j += 256) {
    float v = -INFINITY;
    if (j < t) {
      const float* ej = EQSEQ + ((size_t)b * T_ + j) * Dm;
      float s = 0.f;
      for (int k = 0; k < Dm; ++k) s += ej[k] * eqn[k];
      v = s;
    }
    sc[j] = v;
  }
  const int qn = question[b * T_ + t + 1];
  for (int o = tid; o < 5 * Dm; o += 256) {
    const int i = o / Dm, d = o - i * Dm;
    qsr[i][d] = (i == 0) ? emb_q[(size_t)qn * Dm + d]
                         : emb_s[(size_t)qs_skill_ids[qn * NQNB + i - 1] * Dm + d];
  }
  __syncthreads();
  for (int o = tid; o < 5 * Dm; o += 256) {
    const int i = o / Dm, n = o - i * Dm;
    float acc = qb[n];
    const float* w = qW + (size_t)n * Dm;
    for (int k = 0; k < Dm; ++k) acc += qsr[i][k] * w[k];
    part[i][n] = wW[n] * acc;
  }
  __syncthreads();
  if (tid < 5) {
    float s = 0.f;
    for (int n = 0; n < Dm; ++n) s += part[tid][n];
    SQ[row * 5 + tid] = s;
  }
  if (tid == 0) {
    for (int r = 0; r < RANKK; ++r) {
      float best = -INFINITY; int bi = -1;
      for (int j = 0; j < T_; ++j)
        if (sc[j] > best) { best = sc[j]; bi = j; }
      TOPIDX[row * RANKK + r] = bi;
      if (bi >= 0) sc[bi] = -INFINITY;
    }
  }
}

// ---- sequential LSTM + attention readout, one block per batch row ----
__global__ void __launch_bounds__(256) k_seq(
    const int* __restrict__ question, const int* __restrict__ qs_skill_ids,
    const float* __restrict__ emb_q, const float* __restrict__ emb_s,
    const float* __restrict__ Whh, const float* __restrict__ bhh,
    const float* __restrict__ kW, const float* __restrict__ kb,
    const float* __restrict__ wW, const float* __restrict__ wbp,
    const float* __restrict__ h0, const float* __restrict__ c0,
    const float* __restrict__ GX, const float* __restrict__ SQ,
    const int* __restrict__ TOPIDX, float* __restrict__ out)
{
  __shared__ half_t Hh[T_][Dm];
  __shared__ float SKH[T_];
  __shared__ float hv[Dm], cv[Dm];
  __shared__ float gates[4 * Dm];
  __shared__ float hist[RANKK + 1][Dm];
  __shared__ float qs[5][Dm];
  __shared__ float gmat[5][RANKK + 1];
  __shared__ float partk[Dm];
  const int b = blockIdx.x, tid = threadIdx.x;
  if (tid < Dm) { hv[tid] = h0[b * Dm + tid]; cv[tid] = c0[b * Dm + tid]; }
  if (tid == 0) out[(size_t)b * T_] = 0.5f;
  float skzero = 0.f;
  for (int n = 0; n < Dm; ++n) skzero += wW[Dm + n] * kb[n];
  __syncthreads();
  for (int t = 0; t < Tm1; ++t) {
    const size_t row = (size_t)b * Tm1 + t;
    for (int o = tid; o < 4 * Dm; o += 256) {
      float acc = GX[row * 4 * Dm + o] + bhh[o];
      const float* w = Whh + (size_t)o * Dm;
      for (int k = 0; k < Dm; ++k) acc += w[k] * hv[k];
      gates[o] = acc;
    }
    __syncthreads();
    if (tid < Dm) {
      const float ig = sigmoidf_(gates[tid]);
      const float fg = sigmoidf_(gates[Dm + tid]);
      const float gg = tanhf(gates[2 * Dm + tid]);
      const float og = sigmoidf_(gates[3 * Dm + tid]);
      const float cn = fg * cv[tid] + ig * gg;
      cv[tid] = cn;
      hv[tid] = og * tanhf(cn);
    }
    __syncthreads();
    const int nv = t < RANKK ? t : RANKK;
    for (int o = tid; o < (RANKK + 1) * Dm; o += 256) {
      const int s = o / Dm, d = o - s * Dm;
      float v;
      if (s == 0) v = hv[d];
      else if (s - 1 < nv) v = (float)Hh[TOPIDX[row * RANKK + s - 1]][d];
      else v = 0.f;
      hist[s][d] = v;
    }
    const int qn = question[b * T_ + t + 1];
    for (int o = tid; o < 5 * Dm; o += 256) {
      const int i = o / Dm, d = o - i * Dm;
      qs[i][d] = (i == 0) ? emb_q[(size_t)qn * Dm + d]
                          : emb_s[(size_t)qs_skill_ids[qn * NQNB + i - 1] * Dm + d];
    }
    __syncthreads();
    if (tid < Dm) {
      float acc = kb[tid];
      const float* w = kW + (size_t)tid * Dm;
      for (int k = 0; k < Dm; ++k) acc += w[k] * hv[k];
      partk[tid] = wW[Dm + tid] * acc;
    }
    if (tid >= 128 && tid < 128 + 5 * (RANKK + 1)) {
      const int p = tid - 128, i = p / (RANKK + 1), s = p - i * (RANKK + 1);
      float acc = 0.f;
      for (int k = 0; k < Dm; ++k) acc += qs[i][k] * hist[s][k];
      gmat[i][s] = acc;
    }
    __syncthreads();
    float sk0loc = 0.f;
    if (tid == 0) {
      for (int n = 0; n < Dm; ++n) sk0loc += partk[n];
      const float wb0 = wbp[0];
      float scm[5][RANKK + 1];
      float mx = -INFINITY;
      for (int i = 0; i < 5; ++i) {
        const float sq = SQ[row * 5 + i];
        for (int s = 0; s <= RANKK; ++s) {
          float sco;
          if (s == 0) sco = sq + sk0loc + wb0;
          else if (s - 1 < nv) sco = sq + SKH[TOPIDX[row * RANKK + s - 1]] + wb0;
          else sco = -1e9f;
          scm[i][s] = sco;
          if (sco > mx) mx = sco;
        }
      }
      float den = 0.f, num = 0.f;
      for (int i = 0; i < 5; ++i)
        for (int s = 0; s <= RANKK; ++s) {
          const float e = expf(scm[i][s] - mx);
          den += e; num += e * gmat[i][s];
        }
      out[(size_t)b * T_ + t + 1] = sigmoidf_(num / den);
    }
    __syncthreads();
    if (tid < Dm) Hh[t][tid] = (half_t)((t > 0) ? hv[tid] : 0.f);
    if (tid == 0) SKH[t] = (t > 0) ? sk0loc : skzero;
    __syncthreads();
  }
}

extern "C" void kernel_launch(void* const* d_in, const int* in_sizes, int n_in,
                              void* d_out, int out_size, void* d_ws, size_t ws_size,
                              hipStream_t stream)
{
  (void)in_sizes; (void)n_in; (void)out_size; (void)ws_size;
  const float* emb_q  = (const float*)d_in[0];
  const float* emb_q2 = (const float*)d_in[1];
  const float* emb_s  = (const float*)d_in[2];
  const float* emb_u  = (const float*)d_in[3];
  const float* emb_r  = (const float*)d_in[4];
  const float* w1_q   = (const float*)d_in[5];
  const float* w2_q   = (const float*)d_in[6];
  const float* lin_W  = (const float*)d_in[7];
  const float* lin_b  = (const float*)d_in[8];
  const float* Wih    = (const float*)d_in[9];
  const float* Whh    = (const float*)d_in[10];
  const float* bih    = (const float*)d_in[11];
  const float* bhh    = (const float*)d_in[12];
  const float* agg_W  = (const float*)d_in[13];
  const float* agg_b  = (const float*)d_in[14];
  const float* last_W = (const float*)d_in[15];
  const float* last_b = (const float*)d_in[16];
  const float* qW     = (const float*)d_in[17];
  const float* qb     = (const float*)d_in[18];
  const float* kW     = (const float*)d_in[19];
  const float* kb     = (const float*)d_in[20];
  const float* wW     = (const float*)d_in[21];
  const float* wb     = (const float*)d_in[22];
  const float* h0     = (const float*)d_in[23];
  const float* c0     = (const float*)d_in[24];
  const int* user          = (const int*)d_in[25];
  const int* question      = (const int*)d_in[26];
  const int* response      = (const int*)d_in[27];
  const int* mask          = (const int*)d_in[28];
  const int* q_neighbors   = (const int*)d_in[29];
  const int* s_neighbors   = (const int*)d_in[30];
  const int* u_neighbors   = (const int*)d_in[31];
  const int* q_neighbors_2 = (const int*)d_in[32];
  const int* qs_skill_ids  = (const int*)d_in[33];

  char* ws = (char*)d_ws;
  size_t off = 0;
  auto take = [&](size_t bytes) {
    char* p = ws + off;
    off = (off + bytes + 255) & ~(size_t)255;
    return p;
  };
  half_t* WH2    = (half_t*)take((size_t)112 * Kp * sizeof(half_t));
  half_t* WHLIN  = (half_t*)take((size_t)208 * KC * sizeof(half_t));
  half_t* WHWIH  = (half_t*)take((size_t)400 * KC * sizeof(half_t));
  half_t* XCAT   = (half_t*)take((size_t)MR * KC * sizeof(half_t));
  half_t* XIN    = (half_t*)take((size_t)MR * KC * sizeof(half_t));
  float*  GX     = (float*)take((size_t)MR * 4 * Dm * sizeof(float));
  float*  EQSEQ  = (float*)take((size_t)B_ * T_ * Dm * sizeof(float));
  float*  SQb    = (float*)take((size_t)MR * 5 * sizeof(float));
  int*    TOPIDX = (int*)take((size_t)MR * RANKK * sizeof(int));

  k_pad_weights<<<588, 256, 0, stream>>>(agg_W, lin_W, Wih, WH2, WHLIN, WHWIH);
  k_eqseq<<<7500, 256, 0, stream>>>(question, emb_q, EQSEQ);
  k_zero_h<<<4096, 256, 0, stream>>>(XIN, (size_t)MR * KC);
  k_aggregate<<<dim3(Tm1, B_), 256, 0, stream>>>(
      user, question, response, mask, q_neighbors, s_neighbors,
      u_neighbors, q_neighbors_2, emb_q, emb_q2, emb_s, emb_u, emb_r,
      agg_W, agg_b, last_W, last_b, w1_q, w2_q, WH2, XCAT);
  // x_in = relu(lin(Xcat)) -> f16
  k_gemm<KC><<<dim3(13, MR / 16 / 8), 256, 0, stream>>>(
      XCAT, KC, WHLIN, KC, lin_b, nullptr, XIN, KC, MR, 2 * Dm, 1);
  // gates_x = x_in @ Wih^T + bih -> f32
  k_gemm<KC><<<dim3(25, MR / 16 / 8), 256, 0, stream>>>(
      XIN, KC, WHWIH, KC, bih, GX, nullptr, 4 * Dm, MR, 4 * Dm, 0);
  k_recap<<<dim3(Tm1, B_), 256, 0, stream>>>(
      question, qs_skill_ids, emb_q, emb_s, qW, qb, wW, EQSEQ, SQb, TOPIDX);
  k_seq<<<B_, 256, 0, stream>>>(
      question, qs_skill_ids, emb_q, emb_s, Whh, bhh, kW, kb, wW, wb,
      h0, c0, GX, SQb, TOPIDX, (float*)d_out);
}
